// EdgeHead_47579647705464
// MI455X (gfx1250) — matrix operand order repaired and checked by reading, not compiled, not run on hardware
//
#include <hip/hip_runtime.h>
#include <hip/hip_bf16.h>

// ---- problem constants (match reference) ----
#define N_NODES_C   50000
#define N_EDGES_C   640000
#define HIDDEN_C    128
#define KDIM_C      258            // 2*HIDDEN + EDGE_DIM
#define NCHUNK_C    9              // K padded to 288 = 9 * 32
#define AROW_C      296            // padded A row stride in bf16 (592B: 16B aligned, bank-conflict free)
#define EDGES_PER_STEP 32          // two 16-row M-tiles per step
#define STEPS_C     8
#define EDGES_PER_WG (EDGES_PER_STEP * STEPS_C)   // 256

typedef __bf16 v16bf __attribute__((ext_vector_type(16)));
typedef __bf16 v8bf  __attribute__((ext_vector_type(8)));
typedef float  v8f   __attribute__((ext_vector_type(8)));

// VALU-only 16-lane prefix-sum step: x += dpp_row_shr<N>(x), bound_ctrl -> 0.
// After shr 1,2,4,8 the full 16-lane sum sits in lanes 15 and 31.
#define DPP_ROW_SHR_ADD(x, ctrl)                                               \
    ((x) + __builtin_bit_cast(float,                                           \
         __builtin_amdgcn_update_dpp(0, __builtin_bit_cast(int, (x)),          \
                                     (ctrl), 0xF, 0xF, true)))

__global__ __launch_bounds__(256)
void edge_mlp_wmma_kernel(const float* __restrict__ h,
                          const long long* __restrict__ edge_index,
                          const float* __restrict__ edge_attr,
                          const float* __restrict__ W1,
                          const float* __restrict__ b1,
                          const float* __restrict__ W2,
                          const float* __restrict__ b2,
                          float* __restrict__ out)
{
    // Double-buffered A staging tile: 32 edges x 288(K, padded) bf16
    __shared__ __align__(16) __bf16 Alds[2][EDGES_PER_STEP * AROW_C];
    // Double-buffered cross-wave reduction: [buf][wave][Mtile][lane-half][row-reg]
    __shared__ float red[2][8][2][2][8];

    const int tid  = threadIdx.x;
    const int lane = tid & 31;
    const int wv   = tid >> 5;                 // wave id 0..7 == N col-tile
    const int nB   = wv * 16 + (lane & 15);    // this lane's output column

    // ---------------------------------------------------------------
    // B fragments (W1 col-tile, K padded to 288) in registers.
    // ISA B layout (32x16 bf16): lane = column; lanes 0-15 hold K=c*32+{0..7,16..23},
    // lanes 16-31 hold K=c*32+{8..15,24..31} (matches A layout).
    // ---------------------------------------------------------------
    v16bf bfrag[NCHUNK_C];
    #pragma unroll
    for (int kc = 0; kc < NCHUNK_C; ++kc) {
        const int kbase = kc * 32 + ((lane >> 4) << 3);
        #pragma unroll
        for (int j = 0; j < 16; ++j) {
            const int k = kbase + j + ((j >> 3) << 3);
            const float v = (k < KDIM_C) ? W1[k * HIDDEN_C + nB] : 0.0f;
            bfrag[kc][j] = (__bf16)v;
        }
    }
    const float b1v = b1[nB];
    const float w2v = W2[nB];
    const float b2v = b2[0];

    const int wgBase  = blockIdx.x * EDGES_PER_WG;
    const int m_stage = tid >> 3;   // edge slot (0..31) this thread stages
    const int tsub    = tid & 7;    // 8 threads cooperate per edge row

    // Pipelined gather registers (next step's A data)
    float4 ru[4], rv[4];
    float  attrv = 0.0f;

    // ---- issue gather loads for step s into registers ----
    auto gather = [&](int s) {
        const int e = wgBase + s * EDGES_PER_STEP + m_stage;
        const long long srcL = edge_index[e];
        const long long dstL = edge_index[N_EDGES_C + e];
        const float4* hu4 = (const float4*)(h + (size_t)srcL * HIDDEN_C);
        const float4* hv4 = (const float4*)(h + (size_t)dstL * HIDDEN_C);
        #pragma unroll
        for (int i = 0; i < 4; ++i) {
            ru[i] = hu4[tsub * 4 + i];
            rv[i] = hv4[tsub * 4 + i];
        }
        attrv = (tsub < 2) ? edge_attr[2 * (size_t)e + tsub] : 0.0f;
    };

    // ---- convert staged registers to bf16 and store A tile ----
    auto storeA = [&](int p) {
        __bf16* arow = &Alds[p][m_stage * AROW_C];
        #pragma unroll
        for (int i = 0; i < 2; ++i) {
            const float4 u0 = ru[2 * i], u1 = ru[2 * i + 1];
            const float4 w0 = rv[2 * i], w1 = rv[2 * i + 1];
            v8bf pu, pv;
            pu[0] = (__bf16)u0.x; pu[1] = (__bf16)u0.y; pu[2] = (__bf16)u0.z; pu[3] = (__bf16)u0.w;
            pu[4] = (__bf16)u1.x; pu[5] = (__bf16)u1.y; pu[6] = (__bf16)u1.z; pu[7] = (__bf16)u1.w;
            pv[0] = (__bf16)w0.x; pv[1] = (__bf16)w0.y; pv[2] = (__bf16)w0.z; pv[3] = (__bf16)w0.w;
            pv[4] = (__bf16)w1.x; pv[5] = (__bf16)w1.y; pv[6] = (__bf16)w1.z; pv[7] = (__bf16)w1.w;
            *(v8bf*)&arow[tsub * 16 + i * 8]       = pu;   // K = 0..127   (h_u)
            *(v8bf*)&arow[128 + tsub * 16 + i * 8] = pv;   // K = 128..255 (h_v)
        }
        // K = 256..295: edge_attr then zero pad (8 threads x 5 elements)
        #pragma unroll
        for (int i = 0; i < 5; ++i) {
            const int k = 256 + tsub + 8 * i;
            const float v = (i == 0 && tsub < 2) ? attrv : 0.0f;
            arow[k] = (__bf16)v;
        }
    };

    // ---- finalize step s (reads red[buf], writes 32 logits) ----
    auto finalize = [&](int s, int buf) {
        if (tid < 32) {
            const int tile = tid >> 4;
            const int half = (tid >> 3) & 1;
            const int r    = tid & 7;
            float ssum = b2v;
            #pragma unroll
            for (int w = 0; w < 8; ++w) ssum += red[buf][w][tile][half][r];
            out[wgBase + s * EDGES_PER_STEP + tid] = ssum;
        }
    };

    gather(0);   // prologue

    for (int s = 0; s < STEPS_C; ++s) {
        const int p = s & 1;

        storeA(p);
        __syncthreads();                 // A[p] visible; red[p^1] (step s-1) visible

        if (s > 0) finalize(s - 1, p ^ 1);
        if (s + 1 < STEPS_C) gather(s + 1);   // overlap gather with WMMA below

        // ---- WMMA: 2 M-tiles x 16(N per wave) x 288(K) ----
        v8f acc0 = {}, acc1 = {};
        {
            // ISA A layout (16x32 bf16): lanes 0-15: M=lane, K=c*32+{0..7,16..23};
            // lanes 16-31: M=lane-16, K=c*32+{8..15,24..31}.
            const __bf16* arow0 = &Alds[p][(lane & 15) * AROW_C];
            const __bf16* arow1 = arow0 + 16 * AROW_C;
            const int aoff = ((lane >> 4) << 3);
            #pragma unroll
            for (int kc = 0; kc < NCHUNK_C; ++kc) {
                const v8bf lo0 = *(const v8bf*)&arow0[kc * 32 + aoff];
                const v8bf hi0 = *(const v8bf*)&arow0[kc * 32 + aoff + 16];
                const v8bf lo1 = *(const v8bf*)&arow1[kc * 32 + aoff];
                const v8bf hi1 = *(const v8bf*)&arow1[kc * 32 + aoff + 16];
                v16bf a0, a1;
                #pragma unroll
                for (int j = 0; j < 8; ++j) {
                    a0[j] = lo0[j]; a0[j + 8] = hi0[j];
                    a1[j] = lo1[j]; a1[j + 8] = hi1[j];
                }
                acc0 = __builtin_amdgcn_wmma_f32_16x16x32_bf16(
                           false, a0, false, bfrag[kc], (short)0, acc0, false, false);
                acc1 = __builtin_amdgcn_wmma_f32_16x16x32_bf16(
                           false, a1, false, bfrag[kc], (short)0, acc1, false, false);
            }
        }

        // ---- Fused layer 2: relu(acc + b1[n]) * W2[n]; VALU-only DPP reduce ----
        float p0[8], p1[8];
        #pragma unroll
        for (int r = 0; r < 8; ++r) {
            float h0 = acc0[r] + b1v; h0 = h0 > 0.0f ? h0 : 0.0f;
            float h1 = acc1[r] + b1v; h1 = h1 > 0.0f ? h1 : 0.0f;
            p0[r] = h0 * w2v;
            p1[r] = h1 * w2v;
        }
        #pragma unroll
        for (int r = 0; r < 8; ++r) {
            p0[r] = DPP_ROW_SHR_ADD(p0[r], 0x111);
            p0[r] = DPP_ROW_SHR_ADD(p0[r], 0x112);
            p0[r] = DPP_ROW_SHR_ADD(p0[r], 0x114);
            p0[r] = DPP_ROW_SHR_ADD(p0[r], 0x118);
            p1[r] = DPP_ROW_SHR_ADD(p1[r], 0x111);
            p1[r] = DPP_ROW_SHR_ADD(p1[r], 0x112);
            p1[r] = DPP_ROW_SHR_ADD(p1[r], 0x114);
            p1[r] = DPP_ROW_SHR_ADD(p1[r], 0x118);
        }
        // lane 15 holds sum of lanes 0-15 (edges m=r), lane 31 of lanes 16-31 (m=8+r)
        if (lane == 15) {
            #pragma unroll
            for (int r = 0; r < 8; ++r) {
                red[p][wv][0][0][r] = p0[r];
                red[p][wv][1][0][r] = p1[r];
            }
        }
        if (lane == 31) {
            #pragma unroll
            for (int r = 0; r < 8; ++r) {
                red[p][wv][0][1][r] = p0[r];
                red[p][wv][1][1][r] = p1[r];
            }
        }
    }

    __syncthreads();
    finalize(STEPS_C - 1, (STEPS_C - 1) & 1);
}

extern "C" void kernel_launch(void* const* d_in, const int* in_sizes, int n_in,
                              void* d_out, int out_size, void* d_ws, size_t ws_size,
                              hipStream_t stream) {
    const float*     h  = (const float*)d_in[0];
    const long long* ei = (const long long*)d_in[1];   // int64 edge_index (2, E)
    const float*     ea = (const float*)d_in[2];
    const float*     W1 = (const float*)d_in[3];
    const float*     b1 = (const float*)d_in[4];
    const float*     W2 = (const float*)d_in[5];
    const float*     b2 = (const float*)d_in[6];
    float* outp = (float*)d_out;

    const int n_edges = in_sizes[2] / 2;               // edge_attr is (E, 2)
    const int grid = n_edges / EDGES_PER_WG;           // 640000 / 256 = 2500
    edge_mlp_wmma_kernel<<<grid, 256, 0, stream>>>(h, ei, ea, W1, b1, W2, b2, outp);
}
